// MultiGalerkinNN_84430467104778
// MI455X (gfx1250) — matrix operand order repaired and checked by reading, not compiled, not run on hardware
//
#include <hip/hip_runtime.h>
#include <math.h>

#define B_   16
#define G_   8192
#define UC_  64
#define FCH_ 64
#define IC_  128
#define LDA  40   // padded LDS row stride in halves (32 data + 8 pad, 80B = 16B aligned)

typedef __attribute__((ext_vector_type(16))) _Float16 v16h;
typedef __attribute__((ext_vector_type(8)))  _Float16 v8h;
typedef __attribute__((ext_vector_type(4)))  _Float16 v4h;
typedef __attribute__((ext_vector_type(8)))  float    v8f;
typedef __attribute__((ext_vector_type(4)))  float    v4f;

__device__ __forceinline__ float gelu_f(float x) {
  return 0.5f * x * (1.0f + erff(x * 0.70710678118654752440f));
}

// ---------------- WMMA fragment readers (wave32) ---------------------------
// A tile in LDS: [16][LDA] row-major (m, k). ISA A layout: lane m=lane&15,
// half e -> k = (e>>3)*16 + (e&7) + (lane>=16?8:0)  => two contiguous 8-half runs.
__device__ __forceinline__ v16h frag_a(const _Float16* t, int lane) {
  int m  = lane & 15;
  int kh = (lane >> 4) << 3;
  const _Float16* r = t + m * LDA + kh;
  v8h lo = *(const v8h*)(r);        // k = kh .. kh+7
  v8h hi = *(const v8h*)(r + 16);   // k = 16+kh .. 16+kh+7
  return __builtin_shufflevector(lo, hi, 0, 1, 2, 3, 4, 5, 6, 7,
                                 8, 9, 10, 11, 12, 13, 14, 15);
}
// B tile in LDS TRANSPOSED: [n][LDA] (n, k). ISA B layout: lane n=lane&15,
// half e -> k = (e>>3)*16 + (e&7) + (lane>=16?8:0)  => two contiguous 8-half runs.
__device__ __forceinline__ v16h frag_bT(const _Float16* t, int lane) {
  int n  = lane & 15;
  int kh = (lane >> 4) << 3;
  const _Float16* r = t + n * LDA + kh;
  v8h lo = *(const v8h*)(r);
  v8h hi = *(const v8h*)(r + 16);
  return __builtin_shufflevector(lo, hi, 0, 1, 2, 3, 4, 5, 6, 7,
                                 8, 9, 10, 11, 12, 13, 14, 15);
}
#define WMMA_F16(a, b, c) \
  __builtin_amdgcn_wmma_f32_16x16x32_f16(false, (a), false, (b), (short)0, (c), false, false)

// ---------------- LDS staging helpers --------------------------------------
// A: 16x32 f32 tile (row-major, leading dim ld) -> ta[16][LDA] f16.
__device__ __forceinline__ void stage_a(_Float16* ta, const float* src, size_t ld,
                                        int lane) {
#pragma unroll
  for (int i = 0; i < 4; ++i) {
    int g = lane + 32 * i;
    int r = g >> 3, c4 = (g & 7) << 2;
    v4f v = *(const v4f*)(src + (size_t)r * ld + c4);
    *(v4h*)(ta + r * LDA + c4) = __builtin_convertvector(v, v4h);
  }
}
// B: 32x32 f32 tile, global [k][n] (leading dim ldb) -> tb[32 n][LDA k] f16.
__device__ __forceinline__ void stage_bT_kn32(_Float16* tb, const float* src,
                                              size_t ldb, int lane) {
#pragma unroll
  for (int i = 0; i < 8; ++i) {
    int g = lane + 32 * i;
    int n = g & 31, k = (g >> 5) << 2;
    v4h h;
    h[0] = (_Float16)src[(size_t)(k + 0) * ldb + n];
    h[1] = (_Float16)src[(size_t)(k + 1) * ldb + n];
    h[2] = (_Float16)src[(size_t)(k + 2) * ldb + n];
    h[3] = (_Float16)src[(size_t)(k + 3) * ldb + n];
    *(v4h*)(tb + n * LDA + k) = h;
  }
}
// B: 32x32 f32 tile, global [n][k] (leading dim ldn) -> tb[32 n][LDA k] f16.
__device__ __forceinline__ void stage_bT_nk32(_Float16* tb, const float* src,
                                              size_t ldn, int lane) {
#pragma unroll
  for (int i = 0; i < 8; ++i) {
    int g = lane + 32 * i;
    int n = g >> 3, c4 = (g & 7) << 2;
    v4f v = *(const v4f*)(src + (size_t)n * ldn + c4);
    *(v4h*)(tb + n * LDA + c4) = __builtin_convertvector(v, v4h);
  }
}

// ---------------- fc0: lift a -> u (B,64,G), f (B,64,G) --------------------
__global__ void k_fc0(const float* __restrict__ a,
                      const float* __restrict__ aw, const float* __restrict__ ab,
                      const float* __restrict__ fw, const float* __restrict__ fb,
                      const float* __restrict__ uw, const float* __restrict__ ub,
                      float* __restrict__ u, float* __restrict__ f) {
  int idx = blockIdx.x * blockDim.x + threadIdx.x;
  if (idx >= B_ * G_) return;
  int b = idx / G_, g = idx % G_;
  float a0 = a[(size_t)idx * 2 + 0];
  float a1 = a[(size_t)idx * 2 + 1];
  float fv[64];
#pragma unroll
  for (int j = 0; j < 64; ++j) fv[j] = fw[j] + fw[64 + j] * a1 + fb[j];
  float a2[16];
#pragma unroll
  for (int i = 0; i < 16; ++i) a2[i] = aw[i] * a0 + aw[16 + i] * a1 + ab[i];
  for (int o = 0; o < 64; ++o) {
    float acc = ub[o];
#pragma unroll
    for (int i = 0; i < 16; ++i) acc += a2[i] * uw[i * 64 + o];
#pragma unroll 8
    for (int j = 0; j < 64; ++j) acc += fv[j] * uw[(16 + j) * 64 + o];
    u[((size_t)b * UC_ + o) * G_ + g] = acc;
  }
  for (int j = 0; j < 64; ++j) f[((size_t)b * FCH_ + j) * G_ + g] = fv[j];
}

// ---------------- out = f - conv1d(u, w(64,64,3), pad=1) -------------------
// No __restrict__ on f/out: called in-place (f == out).
__global__ void k_posconv(const float* __restrict__ u, const float* __restrict__ w,
                          const float* f, float* out, int L) {
  int idx = blockIdx.x * blockDim.x + threadIdx.x;
  if (idx >= B_ * UC_ * L) return;
  int x = idx % L, bo = idx / L, o = bo % UC_, b = bo / UC_;
  const float* ubase = u + (size_t)b * UC_ * L;
  float acc = 0.f;
  for (int c = 0; c < UC_; ++c) {
    const float* ur = ubase + (size_t)c * L;
    const float* wr = w + (o * UC_ + c) * 3;
    float s = ur[x] * wr[1];
    if (x > 0)     s += ur[x - 1] * wr[0];
    if (x + 1 < L) s += ur[x + 1] * wr[2];
    acc += s;
  }
  out[idx] = f[idx] - acc;
}

// ---------------- projection: xhat[b,c,m] = sum_x xcat[b,c,x]*wb[x,m] ------
// grid (8, m/32, B), block 32; each wave does a 16x32 output strip.
__global__ void k_proj(const float* __restrict__ u, const float* __restrict__ df,
                       const float* __restrict__ wb, float* __restrict__ xhat,
                       int L, int m) {
  __shared__ __align__(16) _Float16 ta[16 * LDA];
  __shared__ __align__(16) _Float16 tb[32 * LDA];
  int lane = threadIdx.x;
  int c0 = blockIdx.x * 16, n0 = blockIdx.y * 32, b = blockIdx.z;
  const float* Abase = (c0 < UC_) ? (u  + ((size_t)b * UC_ + c0) * L)
                                  : (df + ((size_t)b * UC_ + (c0 - UC_)) * L);
  v8f acc0 = {}, acc1 = {};
  for (int k0 = 0; k0 < L; k0 += 32) {
    stage_a(ta, Abase + k0, L, lane);
    stage_bT_kn32(tb, wb + (size_t)k0 * 128 + n0, 128, lane);
    __syncthreads();
    v16h af = frag_a(ta, lane);
    v16h b0 = frag_bT(tb, lane);
    v16h b1 = frag_bT(tb + 16 * LDA, lane);
    acc0 = WMMA_F16(af, b0, acc0);
    acc1 = WMMA_F16(af, b1, acc1);
    __syncthreads();
  }
  int n = lane & 15, mr0 = (lane >> 4) << 3;
#pragma unroll
  for (int v = 0; v < 8; ++v) {
    size_t row = (size_t)b * IC_ + c0 + mr0 + v;
    xhat[row * m + n0 + n]      = acc0[v];
    xhat[row * m + n0 + 16 + n] = acc1[v];
  }
}

// ---- spectral mul: xhat2[b,o,mm] = sum_i xhat[b,i,mm]*spw[i,o,mm] ---------
// grid (m, 8), block 32. M-rows = batch (exactly 16). Strided gathers (tiny).
__global__ void k_specmul(const float* __restrict__ xhat, const float* __restrict__ spw,
                          float* __restrict__ xhat2, int m) {
  __shared__ __align__(16) _Float16 ta[16 * LDA];
  __shared__ __align__(16) _Float16 tb[16 * LDA];
  int lane = threadIdx.x;
  int mm = blockIdx.x, n0 = blockIdx.y * 16;
  v8f acc = {};
  for (int k0 = 0; k0 < IC_; k0 += 32) {
#pragma unroll
    for (int i = 0; i < 16; ++i) {
      int idx = lane + 32 * i, r = idx >> 5, cc = idx & 31;
      ta[r * LDA + cc] = (_Float16)xhat[((size_t)r * IC_ + k0 + cc) * m + mm];
    }
#pragma unroll
    for (int i = 0; i < 16; ++i) {
      int idx = lane + 32 * i, n = idx & 15, k = idx >> 4;
      tb[n * LDA + k] = (_Float16)spw[((size_t)(k0 + k) * IC_ + n0 + n) * m + mm];
    }
    __syncthreads();
    v16h af = frag_a(ta, lane);
    v16h bf = frag_bT(tb, lane);
    acc = WMMA_F16(af, bf, acc);
    __syncthreads();
  }
  int n = lane & 15, mr0 = (lane >> 4) << 3;
#pragma unroll
  for (int v = 0; v < 8; ++v)
    xhat2[((size_t)(mr0 + v) * IC_ + n0 + n) * m + mm] = acc[v];
}

// ---- 1x1 conv: x1[b,o,x] = w_b[o] + sum_c w_w[o,c]*xcat[b,c,x] ------------
// grid (8, L/32, B), block 32.
__global__ void k_conv1x1(const float* __restrict__ ww, const float* __restrict__ u,
                          const float* __restrict__ df, const float* __restrict__ wbias,
                          float* __restrict__ x1, int L) {
  __shared__ __align__(16) _Float16 ta[16 * LDA];
  __shared__ __align__(16) _Float16 tb[32 * LDA];
  int lane = threadIdx.x;
  int o0 = blockIdx.x * 16, x0 = blockIdx.y * 32, b = blockIdx.z;
  int n = lane & 15, mr0 = (lane >> 4) << 3;
  v8f acc0, acc1;
#pragma unroll
  for (int v = 0; v < 8; ++v) { acc0[v] = wbias[o0 + mr0 + v]; acc1[v] = acc0[v]; }
  for (int k0 = 0; k0 < IC_; k0 += 32) {
    stage_a(ta, ww + (size_t)o0 * IC_ + k0, IC_, lane);
    const float* src = (k0 < UC_) ? (u  + ((size_t)b * UC_ + k0) * L)
                                  : (df + ((size_t)b * UC_ + (k0 - UC_)) * L);
    stage_bT_kn32(tb, src + x0, L, lane);
    __syncthreads();
    v16h af = frag_a(ta, lane);
    v16h b0 = frag_bT(tb, lane);
    v16h b1 = frag_bT(tb + 16 * LDA, lane);
    acc0 = WMMA_F16(af, b0, acc0);
    acc1 = WMMA_F16(af, b1, acc1);
    __syncthreads();
  }
#pragma unroll
  for (int v = 0; v < 8; ++v) {
    size_t row = (size_t)b * IC_ + o0 + mr0 + v;
    x1[row * L + x0 + n]      = acc0[v];
    x1[row * L + x0 + 16 + n] = acc1[v];
  }
}

// ---- reconstruction (accumulating): x1[b,o,x] += sum_mm xhat2[b,o,mm]*bases[x,mm]
// grid (8, L/32, B), block 32. bases is [x][mm] => global [n][k].
__global__ void k_recon(const float* __restrict__ xhat2, const float* __restrict__ bases,
                        float* __restrict__ x1, int L, int m) {
  __shared__ __align__(16) _Float16 ta[16 * LDA];
  __shared__ __align__(16) _Float16 tb[32 * LDA];
  int lane = threadIdx.x;
  int o0 = blockIdx.x * 16, x0 = blockIdx.y * 32, b = blockIdx.z;
  int n = lane & 15, mr0 = (lane >> 4) << 3;
  v8f acc0, acc1;
#pragma unroll
  for (int v = 0; v < 8; ++v) {
    size_t row = (size_t)b * IC_ + o0 + mr0 + v;
    acc0[v] = x1[row * L + x0 + n];
    acc1[v] = x1[row * L + x0 + 16 + n];
  }
  for (int k0 = 0; k0 < m; k0 += 32) {
    stage_a(ta, xhat2 + ((size_t)b * IC_ + o0) * m + k0, m, lane);
    stage_bT_nk32(tb, bases + (size_t)x0 * 128 + k0, 128, lane);
    __syncthreads();
    v16h af = frag_a(ta, lane);
    v16h b0 = frag_bT(tb, lane);
    v16h b1 = frag_bT(tb + 16 * LDA, lane);
    acc0 = WMMA_F16(af, b0, acc0);
    acc1 = WMMA_F16(af, b1, acc1);
    __syncthreads();
  }
#pragma unroll
  for (int v = 0; v < 8; ++v) {
    size_t row = (size_t)b * IC_ + o0 + mr0 + v;
    x1[row * L + x0 + n]      = acc0[v];
    x1[row * L + x0 + 16 + n] = acc1[v];
  }
}

// ---- x1 := xcat + gelu(x1) ------------------------------------------------
__global__ void k_addgelu(const float* __restrict__ u, const float* __restrict__ df,
                          float* x1, int L) {
  int idx = blockIdx.x * blockDim.x + threadIdx.x;
  if (idx >= B_ * IC_ * L) return;
  int x = idx % L, bc = idx / L, c = bc % IC_, b = bc / IC_;
  float xc = (c < UC_) ? u[((size_t)b * UC_ + c) * L + x]
                       : df[((size_t)b * UC_ + (c - UC_)) * L + x];
  x1[idx] = xc + gelu_f(x1[idx]);
}

// ---- usol[b,o,x] = u[b,o,x] + fc_b[o] + sum_c fc_w[o,c]*xg[b,c,x] ---------
// grid (4, L/32, B), block 32. Called in-place (u == usol): each element is
// read only by its own tile's C-init before any store. No restrict on u/usol.
__global__ void k_fc(const float* __restrict__ fcw, const float* __restrict__ xg,
                     const float* u, const float* __restrict__ fcb,
                     float* usol, int L) {
  __shared__ __align__(16) _Float16 ta[16 * LDA];
  __shared__ __align__(16) _Float16 tb[32 * LDA];
  int lane = threadIdx.x;
  int o0 = blockIdx.x * 16, x0 = blockIdx.y * 32, b = blockIdx.z;
  int n = lane & 15, mr0 = (lane >> 4) << 3;
  v8f acc0, acc1;
#pragma unroll
  for (int v = 0; v < 8; ++v) {
    int o = o0 + mr0 + v;
    size_t row = (size_t)b * UC_ + o;
    acc0[v] = u[row * L + x0 + n] + fcb[o];
    acc1[v] = u[row * L + x0 + 16 + n] + fcb[o];
  }
  for (int k0 = 0; k0 < IC_; k0 += 32) {
    stage_a(ta, fcw + (size_t)o0 * IC_ + k0, IC_, lane);
    stage_bT_kn32(tb, xg + ((size_t)b * IC_ + k0) * L + x0, L, lane);
    __syncthreads();
    v16h af = frag_a(ta, lane);
    v16h b0 = frag_bT(tb, lane);
    v16h b1 = frag_bT(tb + 16 * LDA, lane);
    acc0 = WMMA_F16(af, b0, acc0);
    acc1 = WMMA_F16(af, b1, acc1);
    __syncthreads();
  }
#pragma unroll
  for (int v = 0; v < 8; ++v) {
    size_t row = (size_t)b * UC_ + o0 + mr0 + v;
    usol[row * L + x0 + n]      = acc0[v];
    usol[row * L + x0 + 16 + n] = acc1[v];
  }
}

// ---- channel restriction conv: K=3, stride 2, pad 1 -----------------------
__global__ void k_restrict(const float* __restrict__ in, const float* __restrict__ w,
                           float* __restrict__ out, int L) {
  int Lo = L >> 1;
  int idx = blockIdx.x * blockDim.x + threadIdx.x;
  if (idx >= B_ * UC_ * Lo) return;
  int xr = idx % Lo, bo = idx / Lo, o = bo % UC_, b = bo / UC_;
  const float* ib = in + (size_t)b * UC_ * L;
  float acc = 0.f;
  for (int c = 0; c < UC_; ++c) {
    const float* ir = ib + (size_t)c * L;
    const float* wr = w + (o * UC_ + c) * 3;
    int x = 2 * xr;
    if (x > 0)     acc += ir[x - 1] * wr[0];
    acc += ir[x] * wr[1];
    if (x + 1 < L) acc += ir[x + 1] * wr[2];
  }
  out[idx] = acc;
}

// ---- basis restriction: bases (L,128) -> (L/2,128), 1-ch K=3 s2 p1 conv ---
__global__ void k_rbasis(const float* __restrict__ in, const float* __restrict__ w,
                         float* __restrict__ out, int L) {
  int Lo = L >> 1;
  int idx = blockIdx.x * blockDim.x + threadIdx.x;
  if (idx >= Lo * 128) return;
  int j = idx % 128, xr = idx / 128;
  int x = 2 * xr;
  float acc = in[(size_t)x * 128 + j] * w[1];
  if (x > 0)     acc += in[(size_t)(x - 1) * 128 + j] * w[0];
  if (x + 1 < L) acc += in[(size_t)(x + 1) * 128 + j] * w[2];
  out[(size_t)xr * 128 + j] = acc;
}

// ---- out[b,o,x] += convT(uc, p(64,64,4), stride 2, pad 1); in-place on out
__global__ void k_prolong_add(const float* __restrict__ uc, const float* __restrict__ p,
                              float* out, int Lc, int Lo) {
  int idx = blockIdx.x * blockDim.x + threadIdx.x;
  if (idx >= B_ * UC_ * Lo) return;
  int x = idx % Lo, bo = idx / Lo, o = bo % UC_, b = bo / UC_;
  const float* ucb = uc + (size_t)b * UC_ * Lc;
  float acc = out[idx];
#pragma unroll
  for (int k = 0; k < 4; ++k) {
    int t = x + 1 - k;
    if (t >= 0 && (t & 1) == 0) {
      int xi = t >> 1;
      if (xi < Lc) {
        for (int i = 0; i < UC_; ++i)
          acc += ucb[(size_t)i * Lc + xi] * p[(i * UC_ + o) * 4 + k];
      }
    }
  }
  out[idx] = acc;
}

// ---- final head: out[b,g] = gelu(u·W1+b1)·W2 + b2 -------------------------
__global__ void k_fc1(const float* __restrict__ u, const float* __restrict__ w1,
                      const float* __restrict__ b1, const float* __restrict__ w2,
                      const float* __restrict__ b2, float* __restrict__ out) {
  int idx = blockIdx.x * blockDim.x + threadIdx.x;
  if (idx >= B_ * G_) return;
  int g = idx % G_, b = idx / G_;
  float ut[64];
#pragma unroll 8
  for (int c = 0; c < 64; ++c) ut[c] = u[((size_t)b * UC_ + c) * G_ + g];
  float o = b2[0];
  for (int j = 0; j < 128; ++j) {
    float h = b1[j];
#pragma unroll 8
    for (int c = 0; c < 64; ++c) h += ut[c] * w1[c * 128 + j];
    o += gelu_f(h) * w2[j];
  }
  out[idx] = o;
}

extern "C" void kernel_launch(void* const* d_in, const int* in_sizes, int n_in,
                              void* d_out, int out_size, void* d_ws, size_t ws_size,
                              hipStream_t stream) {
  (void)in_sizes; (void)n_in; (void)out_size; (void)ws_size;
  const float* a        = (const float*)d_in[0];
  const float* bases0   = (const float*)d_in[1];
  const float* wbases0  = (const float*)d_in[2];
  const float* fc0_a_w  = (const float*)d_in[3];
  const float* fc0_a_b  = (const float*)d_in[4];
  const float* fc0_f_w  = (const float*)d_in[5];
  const float* fc0_f_b  = (const float*)d_in[6];
  const float* fc0_u_w  = (const float*)d_in[7];
  const float* fc0_u_b  = (const float*)d_in[8];
  const float* pos_w[2] = {(const float*)d_in[9], (const float*)d_in[10]};
  const float* sp_w[3]  = {(const float*)d_in[11], (const float*)d_in[16], (const float*)d_in[21]};
  const float* w_w[3]   = {(const float*)d_in[12], (const float*)d_in[17], (const float*)d_in[22]};
  const float* w_b[3]   = {(const float*)d_in[13], (const float*)d_in[18], (const float*)d_in[23]};
  const float* fc_w[3]  = {(const float*)d_in[14], (const float*)d_in[19], (const float*)d_in[24]};
  const float* fc_b[3]  = {(const float*)d_in[15], (const float*)d_in[20], (const float*)d_in[25]};
  const float* ru[2]    = {(const float*)d_in[26], (const float*)d_in[31]};
  const float* rf[2]    = {(const float*)d_in[27], (const float*)d_in[32]};
  const float* rb[2]    = {(const float*)d_in[28], (const float*)d_in[33]};
  const float* rwb[2]   = {(const float*)d_in[29], (const float*)d_in[34]};
  const float* pw[2]    = {(const float*)d_in[30], (const float*)d_in[35]};
  const float* fc1_w1   = (const float*)d_in[36];
  const float* fc1_b1   = (const float*)d_in[37];
  const float* fc1_w2   = (const float*)d_in[38];
  const float* fc1_b2   = (const float*)d_in[39];

  float* ws = (float*)d_ws;
  size_t off = 0;
  auto alloc = [&](size_t n) { float* p = ws + off; off += n; return p; };
  float* u0  = alloc((size_t)B_ * UC_ * 8192);
  float* f0  = alloc((size_t)B_ * FCH_ * 8192);
  float* u1  = alloc((size_t)B_ * UC_ * 4096);
  float* f1  = alloc((size_t)B_ * FCH_ * 4096);
  float* u2  = alloc((size_t)B_ * UC_ * 2048);
  float* f2  = alloc((size_t)B_ * FCH_ * 2048);
  float* dfb = alloc((size_t)B_ * FCH_ * 8192);
  float* x1  = alloc((size_t)B_ * IC_ * 8192);
  float* xh  = alloc((size_t)B_ * IC_ * 128);
  float* xh2 = alloc((size_t)B_ * IC_ * 128);
  float* bs1 = alloc((size_t)4096 * 128);
  float* wb1 = alloc((size_t)4096 * 128);
  float* bs2 = alloc((size_t)2048 * 128);
  float* wb2 = alloc((size_t)2048 * 128);

  int Ls[3] = {8192, 4096, 2048};
  int ms[3] = {128, 64, 32};
  float* ub[3] = {u0, u1, u2};
  float* fb[3] = {f0, f1, f2};
  const float* bs[3]  = {bases0, bs1, bs2};
  const float* wbs[3] = {wbases0, wb1, wb2};
  float* bsw[3]  = {nullptr, bs1, bs2};
  float* wbsw[3] = {nullptr, wb1, wb2};

  // lift
  k_fc0<<<(B_ * G_ + 255) / 256, 256, 0, stream>>>(a, fc0_a_w, fc0_a_b, fc0_f_w,
                                                   fc0_f_b, fc0_u_w, fc0_u_b, u0, f0);

  for (int l = 0; l < 2; ++l) {
    int L = Ls[l], m = ms[l];
    dim3 ge((B_ * UC_ * L + 255) / 256);
    // df = f - conv(u)
    k_posconv<<<ge, 256, 0, stream>>>(ub[l], pos_w[l], fb[l], dfb, L);
    // galerkin solve (in-place into ub[l])
    k_proj<<<dim3(8, m / 32, B_), 32, 0, stream>>>(ub[l], dfb, wbs[l], xh, L, m);
    k_specmul<<<dim3(m, 8), 32, 0, stream>>>(xh, sp_w[l], xh2, m);
    k_conv1x1<<<dim3(8, L / 32, B_), 32, 0, stream>>>(w_w[l], ub[l], dfb, w_b[l], x1, L);
    k_recon<<<dim3(8, L / 32, B_), 32, 0, stream>>>(xh2, bs[l], x1, L, m);
    k_addgelu<<<(B_ * IC_ * L + 255) / 256, 256, 0, stream>>>(ub[l], dfb, x1, L);
    k_fc<<<dim3(4, L / 32, B_), 32, 0, stream>>>(fc_w[l], x1, ub[l], fc_b[l], ub[l], L);
    // f = f - conv(u_solved)  (in-place)
    k_posconv<<<ge, 256, 0, stream>>>(ub[l], pos_w[l], fb[l], fb[l], L);
    // restrict to next level
    dim3 gr((B_ * UC_ * (L / 2) + 255) / 256);
    k_restrict<<<gr, 256, 0, stream>>>(ub[l], ru[l], ub[l + 1], L);
    k_restrict<<<gr, 256, 0, stream>>>(fb[l], rf[l], fb[l + 1], L);
    dim3 gb(((L / 2) * 128 + 255) / 256);
    k_rbasis<<<gb, 256, 0, stream>>>(bs[l], rb[l], bsw[l + 1], L);
    k_rbasis<<<gb, 256, 0, stream>>>(wbs[l], rwb[l], wbsw[l + 1], L);
  }

  // coarsest level: solver takes f directly
  {
    int L = Ls[2], m = ms[2];
    k_proj<<<dim3(8, m / 32, B_), 32, 0, stream>>>(ub[2], fb[2], wbs[2], xh, L, m);
    k_specmul<<<dim3(m, 8), 32, 0, stream>>>(xh, sp_w[2], xh2, m);
    k_conv1x1<<<dim3(8, L / 32, B_), 32, 0, stream>>>(w_w[2], ub[2], fb[2], w_b[2], x1, L);
    k_recon<<<dim3(8, L / 32, B_), 32, 0, stream>>>(xh2, bs[2], x1, L, m);
    k_addgelu<<<(B_ * IC_ * L + 255) / 256, 256, 0, stream>>>(ub[2], fb[2], x1, L);
    k_fc<<<dim3(4, L / 32, B_), 32, 0, stream>>>(fc_w[2], x1, ub[2], fc_b[2], ub[2], L);
  }

  // upward pass: u_l := u_l + convT(u_{l+1})
  k_prolong_add<<<(B_ * UC_ * 4096 + 255) / 256, 256, 0, stream>>>(ub[2], pw[1], ub[1], 2048, 4096);
  k_prolong_add<<<(B_ * UC_ * 8192 + 255) / 256, 256, 0, stream>>>(ub[1], pw[0], ub[0], 4096, 8192);

  // head
  k_fc1<<<(B_ * G_ + 255) / 256, 256, 0, stream>>>(ub[0], fc1_w1, fc1_b1, fc1_w2,
                                                   fc1_b2, (float*)d_out);
}
